// JSONLSTMEncoder_33990371180854
// MI455X (gfx1250) — matrix operand order repaired and checked by reading, not compiled, not run on hardware
//
#include <hip/hip_runtime.h>

#define D          256
#define NCH        16
#define BATCH      16384
#define WPB        4        // waves per block (128 threads) -> 256 blocks, covers up to 256 WGPs
#define NTHREADS   (WPB * 32)
#define ROWS       16       // batch rows per wave (WMMA M)
#define LDS_STRIDE 132      // dwords per row/lane-slot: 128 data + 4 pad (16B aligned, bank-spread)
#define WF_DWORDS  (256 * LDS_STRIDE)                 // Wf bf16 region
#define FC_DWORDS  (NTHREADS * LDS_STRIDE)            // per-lane fc accumulator region
#define SHMEM_B    ((WF_DWORDS + FC_DWORDS) * 4)      // 202752 bytes

typedef __attribute__((ext_vector_type(4)))  float        v4f;
typedef __attribute__((ext_vector_type(8)))  float        v8f;
typedef __attribute__((ext_vector_type(8)))  __bf16       v8bf;
typedef __attribute__((ext_vector_type(16))) __bf16       v16bf;
typedef __attribute__((ext_vector_type(8)))  unsigned int v8u;

static __device__ __forceinline__ v8f splat8(float x) {
    v8f r;
#pragma unroll
    for (int i = 0; i < 8; ++i) r[i] = x;
    return r;
}

static __device__ __forceinline__ v16bf zero16bf() {
    v16bf r;
#pragma unroll
    for (int i = 0; i < 16; ++i) r[i] = (__bf16)0.0f;
    return r;
}

// two v8f -> one packed bf16 x16 fragment (elements 0..7 = a, 8..15 = b)
static __device__ __forceinline__ v16bf cvt16(v8f a, v8f b) {
    v8bf lo = __builtin_convertvector(a, v8bf);
    v8bf hi = __builtin_convertvector(b, v8bf);
    return __builtin_shufflevector(lo, hi, 0, 1, 2, 3, 4, 5, 6, 7,
                                           8, 9, 10, 11, 12, 13, 14, 15);
}

static __device__ __forceinline__ unsigned int pk2(float a, float b) {
    unsigned short x = __builtin_bit_cast(unsigned short, (__bf16)a);
    unsigned short y = __builtin_bit_cast(unsigned short, (__bf16)b);
    return (unsigned int)x | ((unsigned int)y << 16);
}

// 32B bf16 B-fragment straight from LDS (two ds_load_b128)
static __device__ __forceinline__ v16bf ldsB(const unsigned int* p) {
    v8u w = *(const v8u*)p;
    return __builtin_bit_cast(v16bf, w);
}

// 32B f32 fragment (two 16B-aligned halves) from LDS
static __device__ __forceinline__ v8f ldsF8(const unsigned int* p) {
    v4f a = *(const v4f*)p;
    v4f b = *(const v4f*)(p + 4);
    return __builtin_shufflevector(a, b, 0, 1, 2, 3, 4, 5, 6, 7);
}

static __device__ __forceinline__ void stsF8(unsigned int* p, v8f v) {
    *(v4f*)p       = __builtin_shufflevector(v, v, 0, 1, 2, 3);
    *(v4f*)(p + 4) = __builtin_shufflevector(v, v, 4, 5, 6, 7);
}

// ---- fast transcendentals: gfx1250 V_TANH_F32 if exposed, else exp+rcp ----
static __device__ __forceinline__ float fast_tanh(float x) {
#if __has_builtin(__builtin_amdgcn_tanhf)
    return __builtin_amdgcn_tanhf(x);                 // v_tanh_f32
#else
    float e = __expf(2.0f * x);                       // v_exp_f32
    return 1.0f - 2.0f * __builtin_amdgcn_rcpf(e + 1.0f);   // v_rcp_f32, no IEEE div
#endif
}

static __device__ __forceinline__ float fast_sig(float x) {
    // sigmoid(x) = 0.5 * tanh(x/2) + 0.5  (one TRANS op + FMA on gfx1250)
    return fmaf(0.5f, fast_tanh(0.5f * x), 0.5f);
}

static __device__ __forceinline__ v8f sig8(v8f x) {
    v8f r;
#pragma unroll
    for (int i = 0; i < 8; ++i) r[i] = fast_sig(x[i]);
    return r;
}

static __device__ __forceinline__ v8f tanh8(v8f x) {
    v8f r;
#pragma unroll
    for (int i = 0; i < 8; ++i) r[i] = fast_tanh(x[i]);
    return r;
}

__global__ __launch_bounds__(NTHREADS, 1) void treelstm_wmma_kernel(
    const float* __restrict__ cmem,   // [C, B, D]
    const float* __restrict__ chid,   // [C, B, D]
    const float* __restrict__ Wf,     // [D, D]
    const float* __restrict__ bfv,    // [D]
    const float* __restrict__ Wiou,   // [3D, D]
    const float* __restrict__ biou,   // [3D]
    float* __restrict__ out)          // [2, B, D]
{
    extern __shared__ unsigned int lds[];

    const int tid = threadIdx.x;

    // ---- stage Wf (f32) -> LDS (bf16, row-major, padded stride) -------------
#pragma unroll
    for (int rr = 0; rr < 256 / NTHREADS; ++rr) {
        const int row = tid + rr * NTHREADS;
        const float* wr = Wf + (size_t)row * D;
        unsigned int* lr = lds + row * LDS_STRIDE;
#pragma unroll 8
        for (int i = 0; i < 64; ++i) {
            v4f x = *(const v4f*)(wr + 4 * i);
            lr[2 * i]     = pk2(x[0], x[1]);
            lr[2 * i + 1] = pk2(x[2], x[3]);
        }
    }
    // zero this thread's private fc accumulator slot
    {
        unsigned int* fz = lds + WF_DWORDS + tid * LDS_STRIDE;
#pragma unroll 8
        for (int i = 0; i < LDS_STRIDE; ++i) fz[i] = 0u;
    }
    __syncthreads();

    const int wid   = tid >> 5;
    const int lane  = tid & 31;
    const int ln    = lane & 15;
    const int hi    = lane >> 4;              // K-half / row-half selector
    const int m0    = (blockIdx.x * WPB + wid) * ROWS;
    const int r     = m0 + ln;                // A-fragment row owned by this lane
    const int halfk = hi * 8;                 // K sub-offset used only for prefetch addresses
    const int rhi   = hi * 8;                 // C/D-layout row offset for this lane

    unsigned int* fcb = lds + WF_DWORDS + tid * LDS_STRIDE;   // this lane's fc slots

    v16bf hsum[8];   // bf16 A-fragments of Σ_c h_c, per K-step (stays in VGPRs)
#pragma unroll
    for (int ks = 0; ks < 8; ++ks) hsum[ks] = zero16bf();

    // ================= phase 1: per-child forget GEMM, fused =================
#pragma unroll 1
    for (int c = 0; c < NCH; ++c) {
        const float* hp = chid + ((size_t)c * BATCH + r) * D;

        // software-prefetch next child's streams into cache (global_prefetch_b8)
        if (c + 1 < NCH) {
            const float* hpn = hp + (size_t)BATCH * D;
            const float* mpn = cmem + ((size_t)(c + 1) * BATCH + r) * D;
#pragma unroll
            for (int ks = 0; ks < 8; ++ks) {
                __builtin_prefetch(hpn + ks * 32 + halfk, 0, 3);
                __builtin_prefetch(mpn + ks * 32 + halfk, 0, 3);
            }
        }

        v16bf af[8];
#pragma unroll
        for (int ks = 0; ks < 8; ++ks) {
            const int d0 = ks * 32 + hi * 16;
            v8f a = *(const v8f*)(hp + d0);        // K = d0 .. d0+7
            v8f b = *(const v8f*)(hp + d0 + 8);    // K = d0+8 .. d0+15
            v16bf f = cvt16(a, b);
            af[ks] = f;
            hsum[ks] = hsum[ks] + f;               // v_pk_add_bf16
        }
        const float* mp = cmem + ((size_t)c * BATCH + m0) * D;
#pragma unroll 1
        for (int nt = 0; nt < 16; ++nt) {
            v8f acc = splat8(bfv[nt * 16 + ln]);   // C seeded with bias
            const unsigned int* bp = lds + (nt * 16 + ln) * LDS_STRIDE + hi * 8;
#pragma unroll
            for (int ks = 0; ks < 8; ++ks) {
                v16bf bw = ldsB(bp + ks * 16);
                acc = __builtin_amdgcn_wmma_f32_16x16x32_bf16(
                    false, af[ks], false, bw, (short)0, acc, false, false);
            }
            v8f g = sig8(acc);
            v8f m;
#pragma unroll
            for (int v = 0; v < 8; ++v)
                m[v] = mp[(size_t)(v + rhi) * D + nt * 16 + ln];
            // fc[nt] += g*m   (same-lane LDS rmw; DS ops are in-order per wave)
            unsigned int* fp = fcb + nt * 8;
            v8f old = ldsF8(fp);
            stsF8(fp, old + g * m);
        }
    }

    // ================= phase 2: iou GEMM + gate fusion + store ===============
    float* outm = out;
    float* outh = out + (size_t)BATCH * D;

#pragma unroll 1
    for (int j = 0; j < 16; ++j) {
        v8f accI, accO, accU;
#pragma unroll
        for (int g = 0; g < 3; ++g) {
            const int erow = g * 256 + j * 16 + ln;
            v8f acc = splat8(biou[erow]);
            const float* wp = Wiou + (size_t)erow * D + hi * 16;
#pragma unroll
            for (int ks = 0; ks < 8; ++ks) {
                v8f w0 = *(const v8f*)(wp + ks * 32);
                v8f w1 = *(const v8f*)(wp + ks * 32 + 8);
                v16bf bw = cvt16(w0, w1);
                acc = __builtin_amdgcn_wmma_f32_16x16x32_bf16(
                    false, hsum[ks], false, bw, (short)0, acc, false, false);
            }
            if (g == 0) accI = acc; else if (g == 1) accO = acc; else accU = acc;
        }
        v8f ig = sig8(accI);
        v8f og = sig8(accO);
        v8f ug = tanh8(accU);
        v8f fcj = ldsF8(fcb + j * 8);
        v8f nm = ig * ug + fcj;
        v8f nh = og * tanh8(nm);
#pragma unroll
        for (int v = 0; v < 8; ++v) {
            size_t off = (size_t)(m0 + v + rhi) * D + j * 16 + ln;
            outm[off] = nm[v];
            outh[off] = nh[v];
        }
    }
}

extern "C" void kernel_launch(void* const* d_in, const int* in_sizes, int n_in,
                              void* d_out, int out_size, void* d_ws, size_t ws_size,
                              hipStream_t stream) {
    (void)in_sizes; (void)n_in; (void)out_size; (void)d_ws; (void)ws_size;
    const float* cmem = (const float*)d_in[0];
    const float* chid = (const float*)d_in[1];
    const float* Wf   = (const float*)d_in[2];
    const float* bfv  = (const float*)d_in[3];
    const float* Wiou = (const float*)d_in[4];
    const float* biou = (const float*)d_in[5];
    float* out = (float*)d_out;

    // allow >64KB dynamic LDS (CDNA5 WGP has 320KB); idempotent & capture-safe
    (void)hipFuncSetAttribute((const void*)treelstm_wmma_kernel,
                              hipFuncAttributeMaxDynamicSharedMemorySize, (int)SHMEM_B);

    dim3 grid(BATCH / (WPB * ROWS));   // 256 blocks -> up to 256 WGPs busy
    dim3 block(NTHREADS);              // 4 wave32 waves
    treelstm_wmma_kernel<<<grid, block, (size_t)SHMEM_B, stream>>>(
        cmem, chid, Wf, bfv, Wiou, biou, out);
}
